// Attention_4810363372788
// MI455X (gfx1250) — matrix operand order repaired and checked by reading, not compiled, not run on hardware
//
#include <hip/hip_runtime.h>
#include <hip/hip_bf16.h>
#include <math.h>

#define BB 2
#define SS 2048
#define DD 2048
#define HH 16
#define HD 128

typedef __attribute__((ext_vector_type(16))) __bf16 v16bf;
typedef __attribute__((ext_vector_type(8)))  float  v8f;

__device__ __forceinline__ v8f wmma_bf16(v16bf a, v16bf b, v8f c) {
    return __builtin_amdgcn_wmma_f32_16x16x32_bf16(
        /*neg_a=*/false, a, /*neg_b=*/false, b,
        /*c_mod=*/(short)0, c, /*reuse_a=*/false, /*reuse_b=*/false);
}

// A-fragment (16x32, M x K): lane L -> m = L%16, half = L/16.
// elem e: K = half*16 + e
__device__ __forceinline__ v16bf load_a_frag(const __bf16* row, int half) {
    v16bf a;
#pragma unroll
    for (int e = 0; e < 16; ++e) a[e] = row[half * 16 + e];
    return a;
}

// B-fragment (32x16, K x N): lane L -> n = L%16, half = L/16.
// elem e: K = half*16 + e (16 contiguous bf16 when source is N-major).
__device__ __forceinline__ v16bf load_b_frag(const __bf16* col) {
    v16bf b;
#pragma unroll
    for (int e = 0; e < 16; ++e) b[e] = col[e];
    return b;
}

// ---------------- conversion kernels ----------------
__global__ __launch_bounds__(256) void f32_to_bf16_kernel(const float* __restrict__ in,
                                                          __bf16* __restrict__ out, int n) {
    int i = blockIdx.x * blockDim.x + threadIdx.x;
    if (i < n) out[i] = (__bf16)in[i];
}

// out[c][r] = (bf16) in[r][c] ; in is R x C row-major
__global__ __launch_bounds__(256) void transpose_bf16_kernel(const float* __restrict__ in,
                                                             __bf16* __restrict__ out,
                                                             int R, int C) {
    int i = blockIdx.x * blockDim.x + threadIdx.x;
    if (i >= R * C) return;
    int c = i % C, r = i / C;
    out[(size_t)c * R + r] = (__bf16)in[(size_t)r * C + c];
}

// ---------------- WMMA GEMM: C(f32, MxN) = A(bf16, MxK) * Bt(bf16, NxK)^T ----
// Block: 128 threads = 4 waves; WG tile = 64 rows x 64 cols.
// B tile (32K x 64N, shared by all 4 waves) is staged into LDS with
// double-buffered GLOBAL_LOAD_ASYNC_TO_LDS_B128 (ASYNCcnt pipeline).
// A fragments are register double-buffered.
__global__ __launch_bounds__(128) void gemm_bf16_wmma_kernel(
    const __bf16* __restrict__ A, const __bf16* __restrict__ Bt,
    float* __restrict__ C, int M, int N, int K) {
    __shared__ __bf16 bTile[2][32 * 64];   // [buf][n*32 + k] ; 4KB per buffer

    const int tid  = threadIdx.x;
    const int lane = tid & 31;
    const int wave = tid >> 5;
    const int half = lane >> 4;
    const int ln   = lane & 15;
    const int rowBase = blockIdx.y * 64 + wave * 16;
    const int colBase = blockIdx.x * 64;

    // async-copy mapping: 128 threads cover 64 rows x 2 k-halves (32B each)
    const int cn   = tid >> 1;          // 0..63 : B row (n)
    const int cseg = tid & 1;           // 0..1  : 16-element k segment
    const __bf16* bsrc = Bt + (size_t)(colBase + cn) * K + cseg * 16;
    const unsigned ldsRow = (unsigned)(uintptr_t)(&bTile[0][0]) + cn * 64u + cseg * 32u;
    const unsigned bufStride = 32u * 64u * 2u;

    v8f acc[4] = {v8f{}, v8f{}, v8f{}, v8f{}};
    const __bf16* arow = A + (size_t)(rowBase + ln) * K;

    // prologue: issue async copies for k0 = 0 into buffer 0
    {
        unsigned long long g = (unsigned long long)(uintptr_t)bsrc;
        asm volatile(
            "global_load_async_to_lds_b128 %0, %1, off\n\t"
            "global_load_async_to_lds_b128 %0, %1, off offset:16"
            :: "v"(ldsRow), "v"(g) : "memory");
    }
    v16bf a = load_a_frag(arow, half);

    int buf = 0;
    for (int k0 = 0; k0 < K; k0 += 32) {
        const int k1 = (k0 + 32 < K) ? (k0 + 32) : 0;   // clamp: harmless re-copy
        // issue async copies for next k-chunk into the other buffer
        {
            unsigned long long g = (unsigned long long)(uintptr_t)(bsrc + k1);
            unsigned l = ldsRow + (buf ^ 1) * bufStride;
            asm volatile(
                "global_load_async_to_lds_b128 %0, %1, off\n\t"
                "global_load_async_to_lds_b128 %0, %1, off offset:16"
                :: "v"(l), "v"(g) : "memory");
        }
        // prefetch next A fragment into registers
        v16bf a_n = load_a_frag(arow + k1, half);
        // wait for current buffer's 2 async copies (next buffer's 2 stay in flight)
        asm volatile("s_wait_asynccnt 0x2" ::: "memory");
        __syncthreads();
        // consume current buffer
#pragma unroll
        for (int nt = 0; nt < 4; ++nt) {
            const __bf16* bp = &bTile[buf][(nt * 16 + ln) * 32 + half * 16];
            v16bf b = load_b_frag(bp);
            acc[nt] = wmma_bf16(a, b, acc[nt]);
        }
        __syncthreads();   // everyone done reading before buf is overwritten
        a = a_n;
        buf ^= 1;
    }
#pragma unroll
    for (int nt = 0; nt < 4; ++nt) {
#pragma unroll
        for (int r = 0; r < 8; ++r) {
            int row = rowBase + r + 8 * half;
            int col = colBase + nt * 16 + ln;
            C[(size_t)row * N + col] = acc[nt][r];
        }
    }
}

// ---------------- RoPE + repack -------------------------------------------
// qkv f32 [B,S,3,H,HD] -> Q,K bf16 [B,H,S,HD] (roped), Vt bf16 [B,H,HD,S]
__global__ __launch_bounds__(256) void rope_pack_kernel(
    const float* __restrict__ qkv, __bf16* __restrict__ Q,
    __bf16* __restrict__ Kd, __bf16* __restrict__ Vt) {
    int idx = blockIdx.x * blockDim.x + threadIdx.x;
    const int total = BB * SS * HH * (HD / 2);
    if (idx >= total) return;
    int i = idx % (HD / 2); int t = idx / (HD / 2);
    int h = t % HH; t /= HH;
    int s = t % SS; int b = t / SS;

    const size_t rowbase = ((size_t)b * SS + s) * (3 * HH * HD);
    const size_t qoff = rowbase + 0 * HH * HD + h * HD;
    const size_t koff = rowbase + 1 * HH * HD + h * HD;
    const size_t voff = rowbase + 2 * HH * HD + h * HD;
    float q1 = qkv[qoff + i], q2 = qkv[qoff + i + HD / 2];
    float k1 = qkv[koff + i], k2 = qkv[koff + i + HD / 2];
    float v1 = qkv[voff + i], v2 = qkv[voff + i + HD / 2];

    float freq = __expf(-logf(10000.0f) * (float)i * (1.0f / (HD / 2)));
    float ang = (float)s * freq;
    float c = __cosf(ang), sn = __sinf(ang);

    size_t qo = (((size_t)b * HH + h) * SS + s) * HD;
    Q[qo + i]           = (__bf16)(q1 * c - q2 * sn);
    Q[qo + i + HD / 2]  = (__bf16)(q2 * c + q1 * sn);
    Kd[qo + i]          = (__bf16)(k1 * c - k2 * sn);
    Kd[qo + i + HD / 2] = (__bf16)(k2 * c + k1 * sn);

    size_t vtbase = ((size_t)b * HH + h) * (size_t)HD * SS;
    Vt[vtbase + (size_t)i * SS + s]            = (__bf16)v1;
    Vt[vtbase + (size_t)(i + HD / 2) * SS + s] = (__bf16)v2;
}

// ---------------- Flash attention (one wave per 16 query rows) ------------
// Q,K: [B,H,S,HD] bf16 ; Vt: [B,H,HD,S] bf16 ; O: [B,S,H*HD] bf16
__global__ __launch_bounds__(32) void attn_wmma_kernel(
    const __bf16* __restrict__ Q, const __bf16* __restrict__ Kd,
    const __bf16* __restrict__ Vt, __bf16* __restrict__ O) {
    __shared__ __bf16 pShared[16 * 64];

    const int lane = threadIdx.x;
    const int half = lane >> 4;
    const int ln   = lane & 15;
    const int qt = blockIdx.x;   // S/16 tiles
    const int h  = blockIdx.y;
    const int b  = blockIdx.z;

    const size_t head = ((size_t)b * HH + h) * (size_t)SS * HD;
    const __bf16* Qh = Q  + head;
    const __bf16* Kh = Kd + head;
    const __bf16* Vh = Vt + head;   // [HD][S]

    v16bf qf[4];
    const __bf16* qrow = Qh + (size_t)(qt * 16 + ln) * HD;
#pragma unroll
    for (int kc = 0; kc < 4; ++kc) qf[kc] = load_a_frag(qrow + kc * 32, half);

    float m_i[8], l_i[8];
#pragma unroll
    for (int r = 0; r < 8; ++r) { m_i[r] = -3.0e38f; l_i[r] = 0.0f; }
    v8f acc[8] = {v8f{}, v8f{}, v8f{}, v8f{}, v8f{}, v8f{}, v8f{}, v8f{}};

    const float scale = 0.08838834764831845f;  // 1/sqrt(128)
    const int nkb = (qt * 16 + 15) / 64 + 1;   // causal key blocks of 64

    for (int kb = 0; kb < nkb; ++kb) {
        // scores: 16 q x 64 keys
        v8f sc[4] = {v8f{}, v8f{}, v8f{}, v8f{}};
#pragma unroll
        for (int nt = 0; nt < 4; ++nt) {
            const int key = kb * 64 + nt * 16 + ln;
            const __bf16* krow = Kh + (size_t)key * HD;
            v16bf kf[4];
#pragma unroll
            for (int kc = 0; kc < 4; ++kc)
                kf[kc] = load_b_frag(krow + kc * 32 + half * 16);
#pragma unroll
            for (int kc = 0; kc < 4; ++kc)
                sc[nt] = wmma_bf16(qf[kc], kf[kc], sc[nt]);
        }
        // scale + causal mask
#pragma unroll
        for (int nt = 0; nt < 4; ++nt) {
            const int key = kb * 64 + nt * 16 + ln;
#pragma unroll
            for (int r = 0; r < 8; ++r) {
                const int row = qt * 16 + r + 8 * half;
                float s = sc[nt][r] * scale;
                sc[nt][r] = (key <= row) ? s : -1.0e30f;
            }
        }
        // online softmax (rows live per half-wave; 16-lane shuffle reductions)
        float mnew[8], ladd[8];
#pragma unroll
        for (int r = 0; r < 8; ++r) {
            float v = fmaxf(fmaxf(sc[0][r], sc[1][r]), fmaxf(sc[2][r], sc[3][r]));
#pragma unroll
            for (int m = 1; m < 16; m <<= 1) v = fmaxf(v, __shfl_xor(v, m, 32));
            mnew[r] = fmaxf(m_i[r], v);
        }
#pragma unroll
        for (int r = 0; r < 8; ++r) {
            float lsum = 0.0f;
#pragma unroll
            for (int nt = 0; nt < 4; ++nt) {
                float p = __expf(sc[nt][r] - mnew[r]);
                sc[nt][r] = p;
                lsum += p;
            }
#pragma unroll
            for (int m = 1; m < 16; m <<= 1) lsum += __shfl_xor(lsum, m, 32);
            ladd[r] = lsum;
        }
#pragma unroll
        for (int r = 0; r < 8; ++r) {
            float rs = __expf(m_i[r] - mnew[r]);
            l_i[r] = l_i[r] * rs + ladd[r];
            m_i[r] = mnew[r];
#pragma unroll
            for (int t = 0; t < 8; ++t) acc[t][r] *= rs;
        }
        // C-layout P -> LDS -> A-layout fragments
#pragma unroll
        for (int nt = 0; nt < 4; ++nt)
#pragma unroll
            for (int r = 0; r < 8; ++r)
                pShared[(r + 8 * half) * 64 + nt * 16 + ln] = (__bf16)sc[nt][r];
        __syncthreads();
        // out += P(16x64) * V(64x128)
#pragma unroll
        for (int kc2 = 0; kc2 < 2; ++kc2) {
            v16bf pf = load_a_frag(&pShared[ln * 64 + kc2 * 32], half);
#pragma unroll
            for (int hg = 0; hg < 2; ++hg) {
                v16bf vf[4];
#pragma unroll
                for (int j = 0; j < 4; ++j) {
                    const int ht = hg * 4 + j;
                    vf[j] = load_b_frag(Vh + (size_t)(ht * 16 + ln) * SS +
                                        kb * 64 + kc2 * 32 + half * 16);
                }
#pragma unroll
                for (int j = 0; j < 4; ++j)
                    acc[hg * 4 + j] = wmma_bf16(pf, vf[j], acc[hg * 4 + j]);
            }
        }
        __syncthreads();
    }
    // epilogue: normalize, store bf16 [B,S,H*HD]
#pragma unroll
    for (int ht = 0; ht < 8; ++ht) {
#pragma unroll
        for (int r = 0; r < 8; ++r) {
            const int row = qt * 16 + r + 8 * half;
            const int col = h * HD + ht * 16 + ln;
            O[((size_t)b * SS + row) * (HH * HD) + col] = (__bf16)(acc[ht][r] / l_i[r]);
        }
    }
}

extern "C" void kernel_launch(void* const* d_in, const int* in_sizes, int n_in,
                              void* d_out, int out_size, void* d_ws, size_t ws_size,
                              hipStream_t stream) {
    (void)in_sizes; (void)n_in; (void)out_size; (void)ws_size;
    const float* x     = (const float*)d_in[0];  // [B,S,D]
    const float* w_qkv = (const float*)d_in[1];  // [D, 3*H*HD]
    const float* w_out = (const float*)d_in[2];  // [H*HD, D]
    float* out = (float*)d_out;                  // [B,S,D]

    const int M  = BB * SS;          // 4096
    const int N1 = 3 * HH * HD;      // 6144
    const int K1 = DD;               // 2048

    char* ws = (char*)d_ws;
    size_t off = 0;
    __bf16* xbf   = (__bf16*)(ws + off); off += (size_t)M * K1 * 2;
    __bf16* wqkvT = (__bf16*)(ws + off); off += (size_t)N1 * K1 * 2;
    __bf16* woutT = (__bf16*)(ws + off); off += (size_t)DD * (HH * HD) * 2;
    float*  qkvf  = (float*) (ws + off); off += (size_t)M * N1 * 4;
    __bf16* Qb    = (__bf16*)(ws + off); off += (size_t)BB * HH * SS * HD * 2;
    __bf16* Kb    = (__bf16*)(ws + off); off += (size_t)BB * HH * SS * HD * 2;
    __bf16* Vtb   = (__bf16*)(ws + off); off += (size_t)BB * HH * HD * SS * 2;
    __bf16* attnO = (__bf16*)(ws + off); off += (size_t)M * (HH * HD) * 2;

    { // 1) x -> bf16
        int n = M * K1;
        f32_to_bf16_kernel<<<(n + 255) / 256, 256, 0, stream>>>(x, xbf, n);
    }
    { // 2) w_qkv [K1 x N1] -> wqkvT [N1 x K1] bf16
        int n = K1 * N1;
        transpose_bf16_kernel<<<(n + 255) / 256, 256, 0, stream>>>(w_qkv, wqkvT, K1, N1);
    }
    { // 3) w_out [H*HD x D] -> woutT [D x H*HD] bf16
        int n = (HH * HD) * DD;
        transpose_bf16_kernel<<<(n + 255) / 256, 256, 0, stream>>>(w_out, woutT, HH * HD, DD);
    }
    { // 4) qkv = x @ w_qkv  (bf16 WMMA, f32 accum)
        dim3 grid(N1 / 64, M / 64);
        gemm_bf16_wmma_kernel<<<grid, 128, 0, stream>>>(xbf, wqkvT, qkvf, M, N1, K1);
    }
    { // 5) RoPE + repack
        int n = BB * SS * HH * (HD / 2);
        rope_pack_kernel<<<(n + 255) / 256, 256, 0, stream>>>(qkvf, Qb, Kb, Vtb);
    }
    { // 6) causal flash attention
        dim3 grid(SS / 16, HH, BB);
        attn_wmma_kernel<<<grid, 32, 0, stream>>>(Qb, Kb, Vtb, attnO);
    }
    { // 7) out = attnO @ w_out
        dim3 grid(DD / 64, M / 64);
        gemm_bf16_wmma_kernel<<<grid, 128, 0, stream>>>(attnO, woutT, out, M, DD, HH * HD);
    }
}
